// NeuromorphicNetwork_51805895524797
// MI455X (gfx1250) — matrix operand order repaired and checked by reading, not compiled, not run on hardware
//
#include <hip/hip_runtime.h>

// Persistent fused SNN kernel for MI455X (gfx1250, wave32, WMMA).
// One workgroup = 16 batch rows, 16 waves. Weights W1/W2 resident in VGPRs
// as WMMA B-fragments; W3^T staged into LDS via CDNA5 async-to-LDS DMA;
// spikes/rates live in bank-conflict-free padded LDS buffers; membrane
// potentials live in the fp32 WMMA accumulator registers.

typedef __attribute__((ext_vector_type(16))) _Float16 v16h;
typedef __attribute__((ext_vector_type(8)))  _Float16 v8h;
typedef __attribute__((ext_vector_type(8)))  float    v8f;

#define D_DIM   256
#define H_DIM   256
#define A_DIM   64
#define M_TILE  16
#define NTHREADS 512
#define SIM_T   100
#define DECAY   0.95122942450071400910f   // exp(-1/20)
#define ONEMD   0.04877057549928599090f   // 1 - exp(-1/20)

// Row stride (in halfs) for LDS matrices: 264 halfs = 528 B. 528 mod 256 = 16,
// so lane m's 16B access lands at bank-group m*4: 16 lanes cover all 64 banks
// exactly once -> conflict-free b128 reads.
#define BUF_STRIDE 264

// Hard scheduling fence: keeps the one-time prologue loads from being fused
// into one giant region that blows the VGPR budget (round-2 showed spills).
#define SCHED_FENCE() __builtin_amdgcn_sched_barrier(0)

#define WMMA_F16(a, b, c) \
  __builtin_amdgcn_wmma_f32_16x16x32_f16(false, (a), false, (b), (short)0, (c), false, false)

// A-fragment loader from a row-major [16][BUF_STRIDE] f16 LDS buffer.
// Layout (16-bit A 16x32): lanes 0-15 hold row m=lane, K=[k0..k0+15];
// lanes 16-31 hold K=[k0+16..k0+31] (contiguous, matching B fragments).
__device__ __forceinline__ v16h load_a_lds(const _Float16* buf, int lmod, int lhalf, int c) {
  const _Float16* p = buf + lmod * BUF_STRIDE + c * 32 + lhalf * 16;
  v8h lo = *(const v8h*)(p);
  v8h hi = *(const v8h*)(p + 8);
  return __builtin_shufflevector(lo, hi, 0,1,2,3,4,5,6,7,8,9,10,11,12,13,14,15);
}

__global__ __launch_bounds__(NTHREADS, 1)
void snn_persistent_kernel(const float* __restrict__ x,
                           const float* __restrict__ encW,   // [D,D] row-major [out,in]
                           const float* __restrict__ encB,   // [D]
                           const float* __restrict__ W1,     // [D,H]
                           const float* __restrict__ W2,     // [H,H]
                           const float* __restrict__ W3,     // [H,A]
                           float* __restrict__ out)          // [B,A]
{
  __shared__ _Float16 lds_rates[M_TILE * BUF_STRIDE];  // 8.25 KB
  __shared__ _Float16 lds_s1[M_TILE * BUF_STRIDE];     // 8.25 KB
  __shared__ _Float16 lds_s2[M_TILE * BUF_STRIDE];     // 8.25 KB
  __shared__ _Float16 lds_w3t[A_DIM * BUF_STRIDE];     // 33 KB, W3 transposed [n][k]
  __shared__ float    lds_stage[1024];                 // 4 KB async staging chunk

  const int tid   = threadIdx.x;
  const int lane  = tid & 31;
  const int wave  = tid >> 5;           // 0..15
  const int lmod  = lane & 15;
  const int lhalf = lane >> 4;          // 0 or 1
  const int b0    = blockIdx.x * M_TILE;
  const int nb    = wave * 16;          // this wave's N-tile base (layers 1/2, encoder)

  __builtin_prefetch(W1, 0, 0);
  __builtin_prefetch(W2, 0, 0);
  __builtin_prefetch(W3, 0, 0);

  // ---- Phase 1: stage W3^T into LDS via CDNA5 async global->LDS DMA
  // (ASYNCcnt path). 16 chunks of 4 KB raw fp32, then convert+transpose into
  // the padded f16 W3^T buffer.
  {
    const unsigned stage_base = (unsigned)(uintptr_t)(void*)lds_stage;
    for (int it = 0; it < 16; ++it) {
      if (tid < 256) {   // waves 0..7, wave-uniform
        unsigned      ldsoff = stage_base + (unsigned)tid * 16u;
        unsigned long gaddr  = (unsigned long)(uintptr_t)W3
                             + (unsigned long)it * 4096ul + (unsigned long)tid * 16ul;
        asm volatile("global_load_async_to_lds_b128 %0, %1, off"
                     :: "v"(ldsoff), "v"(gaddr) : "memory");
      }
      asm volatile("s_wait_asynccnt 0x0" ::: "memory");
      __syncthreads();
      // convert / transpose this chunk: flat f = k*64 + n  ->  w3t[n][k]
      for (int j = tid; j < 1024; j += NTHREADS) {
        int f = it * 1024 + j;
        int k = f >> 6;
        int n = f & 63;
        lds_w3t[n * BUF_STRIDE + k] = (_Float16)lds_stage[j];
      }
      __syncthreads();
    }
  }

  // ---- Phase 2: encoder rates = sigmoid(x @ encW^T + encB), cols nb..nb+15.
  // Runs BEFORE the resident-weight fill so its temporaries don't coexist
  // with the 128 weight VGPRs.
  {
    v8f cacc = {};
#pragma unroll
    for (int c = 0; c < 8; ++c) {
      // A-fragment from x (fp32 global -> f16)
      const int koff = c * 32 + lhalf * 16;
      const float* gx = x + (size_t)(b0 + lmod) * D_DIM + koff;
      v16h a;
#pragma unroll
      for (int j = 0; j < 8; ++j) {
        a[j]     = (_Float16)gx[j];
        a[j + 8] = (_Float16)gx[8 + j];
      }
      // B-fragment: B[k][o] = encW[o][k] -> contiguous along k for fixed o
      const int kb = c * 32 + lhalf * 16;
      const float* gw = encW + (size_t)(nb + lmod) * D_DIM + kb;
      v16h b;
#pragma unroll
      for (int j = 0; j < 16; ++j) b[j] = (_Float16)gw[j];
      cacc = WMMA_F16(a, b, cacc);
      SCHED_FENCE();   // limit in-flight load live-ranges to one k-chunk
    }
    const float bias = encB[nb + lmod];
    const int cg = nb + lmod;
#pragma unroll
    for (int r = 0; r < 8; ++r) {
      float z = cacc[r] + bias;
      float s = 1.0f / (1.0f + __expf(-z));
      lds_rates[(lhalf * 8 + r) * BUF_STRIDE + cg] = (_Float16)s;
    }
  }
  SCHED_FENCE();
  __syncthreads();

  // ---- Phase 3: fill resident B-fragments for W1, then W2 (16 cols each).
  // B layout: lane holds column n=lmod, 16 contiguous K halves at kb.
  v16h bw1[8], bw2[8];
#pragma unroll
  for (int c = 0; c < 8; ++c) {
    const int kb  = c * 32 + lhalf * 16;
    const int col = nb + lmod;
    v16h f1;
#pragma unroll
    for (int j = 0; j < 16; ++j)
      f1[j] = (_Float16)W1[(size_t)(kb + j) * H_DIM + col];
    bw1[c] = f1;
    SCHED_FENCE();   // one 16-load chunk in flight at a time
  }
#pragma unroll
  for (int c = 0; c < 8; ++c) {
    const int kb  = c * 32 + lhalf * 16;
    const int col = nb + lmod;
    v16h f2;
#pragma unroll
    for (int j = 0; j < 16; ++j)
      f2[j] = (_Float16)W2[(size_t)(kb + j) * H_DIM + col];
    bw2[c] = f2;
    SCHED_FENCE();
  }

  // ---- Persistent LIF state in registers ----
  float v1s[8], v2s[8], v3s[8], accs[8];
#pragma unroll
  for (int r = 0; r < 8; ++r) { v1s[r] = 0.f; v2s[r] = 0.f; v3s[r] = 0.f; accs[r] = 0.f; }

  const int colg = nb + lmod;
  const int nb3  = (wave & 3) * 16;
  const int col3 = nb3 + lmod;

  for (int t = 0; t < SIM_T; ++t) {
    // ---- Layer 1: syn = rates @ W1 ----
    {
      v8f c8 = {};
#pragma unroll
      for (int c = 0; c < 8; ++c) {
        v16h a = load_a_lds(lds_rates, lmod, lhalf, c);
        c8 = WMMA_F16(a, bw1[c], c8);
      }
#pragma unroll
      for (int r = 0; r < 8; ++r) {
        float v  = v1s[r] * DECAY + c8[r] * ONEMD;
        float sp = (v >= 1.0f) ? 1.0f : 0.0f;
        v1s[r]   = (v >= 1.0f) ? 0.0f : v;
        lds_s1[(lhalf * 8 + r) * BUF_STRIDE + colg] = (_Float16)sp;
      }
    }
    __syncthreads();

    // ---- Layer 2: syn = s1 @ W2 ----
    {
      v8f c8 = {};
#pragma unroll
      for (int c = 0; c < 8; ++c) {
        v16h a = load_a_lds(lds_s1, lmod, lhalf, c);
        c8 = WMMA_F16(a, bw2[c], c8);
      }
#pragma unroll
      for (int r = 0; r < 8; ++r) {
        float v  = v2s[r] * DECAY + c8[r] * ONEMD;
        float sp = (v >= 1.0f) ? 1.0f : 0.0f;
        v2s[r]   = (v >= 1.0f) ? 0.0f : v;
        lds_s2[(lhalf * 8 + r) * BUF_STRIDE + colg] = (_Float16)sp;
      }
    }
    __syncthreads();

    // ---- Layer 3: syn = s2 @ W3 (waves 0..3 own the 64 output cols) ----
    if (wave < 4) {                       // wave-uniform branch: EXEC all-ones inside
      v8f c8 = {};
#pragma unroll
      for (int c = 0; c < 8; ++c) {
        v16h a = load_a_lds(lds_s2, lmod, lhalf, c);
        const int kb = c * 32 + lhalf * 16;
        const _Float16* p = lds_w3t + (nb3 + lmod) * BUF_STRIDE + kb;
        v8h lo = *(const v8h*)(p);
        v8h hi = *(const v8h*)(p + 8);
        v16h b = __builtin_shufflevector(lo, hi, 0,1,2,3,4,5,6,7,8,9,10,11,12,13,14,15);
        c8 = WMMA_F16(a, b, c8);
      }
#pragma unroll
      for (int r = 0; r < 8; ++r) {
        float v  = v3s[r] * DECAY + c8[r] * ONEMD;
        float sp = (v >= 1.0f) ? 1.0f : 0.0f;
        v3s[r]   = (v >= 1.0f) ? 0.0f : v;
        accs[r] += sp;
      }
    }
    __syncthreads();
  }

  // ---- Write output: acc / SIM_T ----
  if (wave < 4) {
#pragma unroll
    for (int r = 0; r < 8; ++r) {
      out[(size_t)(b0 + lhalf * 8 + r) * A_DIM + col3] = accs[r] * (1.0f / (float)SIM_T);
    }
  }
}

extern "C" void kernel_launch(void* const* d_in, const int* in_sizes, int n_in,
                              void* d_out, int out_size, void* d_ws, size_t ws_size,
                              hipStream_t stream) {
  const float* x    = (const float*)d_in[0];
  const float* encW = (const float*)d_in[1];
  const float* encB = (const float*)d_in[2];
  const float* W1   = (const float*)d_in[3];
  const float* W2   = (const float*)d_in[4];
  const float* W3   = (const float*)d_in[5];
  float* out = (float*)d_out;

  const int B = in_sizes[0] / D_DIM;      // 8192
  dim3 grid(B / M_TILE);                  // 512 workgroups
  dim3 block(NTHREADS);                   // 16 waves of 32
  hipLaunchKernelGGL(snn_persistent_kernel, grid, block, 0, stream,
                     x, encW, encB, W1, W2, W3, out);
}
